// GATv2Encoder_65910568124788
// MI455X (gfx1250) — matrix operand order repaired and checked by reading, not compiled, not run on hardware
//
#include <hip/hip_runtime.h>
#include <hip/hip_bf16.h>

typedef __attribute__((ext_vector_type(16))) _Float16 v16h;
typedef __attribute__((ext_vector_type(8)))  _Float16 v8h;
typedef __attribute__((ext_vector_type(8)))  float    v8f;

#define DCH 64   // both layers have 64 output channels

// ---------------- helpers ----------------
__device__ __forceinline__ unsigned fmap(float f) {
    unsigned u = __float_as_uint(f);
    return (u & 0x80000000u) ? ~u : (u | 0x80000000u);
}
__device__ __forceinline__ float funmap(unsigned u) {
    return __uint_as_float((u & 0x80000000u) ? (u ^ 0x80000000u) : ~u);
}
__device__ __forceinline__ float lrelu(float v) { return v > 0.f ? v : 0.2f * v; }
__device__ __forceinline__ float elu(float v)   { return v > 0.f ? v : __expf(v) - 1.f; }

// ---------------- weight pack: W[K x 64] f32 -> Wt[64 x K] f16 ----------------
__global__ void pack_wt_kernel(const float* __restrict__ W, int Kdim, _Float16* __restrict__ Wt) {
    int i = blockIdx.x * blockDim.x + threadIdx.x;
    if (i >= Kdim * DCH) return;
    int k = i / DCH, d = i - k * DCH;
    Wt[(size_t)d * Kdim + k] = (_Float16)W[i];
}

__global__ void cvt16_kernel(const float* __restrict__ x, _Float16* __restrict__ y, size_t n) {
    size_t i = (size_t)blockIdx.x * blockDim.x + threadIdx.x;
    if (i < n) y[i] = (_Float16)x[i];
}

// ---------------- dual GEMM via WMMA: fs = A@Wsrc + bs, fd = A@Wdst + bd ----------------
// A: [Nrows x Kdim] f16 row-major.  WsT/WdT: [64 x Kdim] f16 (transposed weights).
// One wave -> 16 rows x 64 cols for BOTH outputs (A fragment reused across 8 WMMAs/K-step).
__global__ void __launch_bounds__(256)
gemm_dual_wmma_kernel(const _Float16* __restrict__ A, int Kdim, int Nrows,
                      const _Float16* __restrict__ WsT, const _Float16* __restrict__ WdT,
                      const float* __restrict__ bs, const float* __restrict__ bd,
                      float* __restrict__ fs, float* __restrict__ fd) {
    const int lane = threadIdx.x & 31;
    const int wave = threadIdx.x >> 5;
    const int tileM = blockIdx.x * 8 + wave;
    const int rowBase = tileM * 16;
    if (rowBase >= Nrows) return;           // uniform per wave

    const int  l15 = lane & 15;
    const bool hi  = lane >= 16;
    int arow = rowBase + l15;
    if (arow >= Nrows) arow = Nrows - 1;    // clamp loads; stores predicated below
    const _Float16* Arow = A + (size_t)arow * Kdim + (hi ? 16 : 0);

    v8f accS[4] = {}, accD[4] = {};
    for (int k = 0; k < Kdim; k += 32) {
        v8h a0 = *(const v8h*)(Arow + k);
        v8h a1 = *(const v8h*)(Arow + k + 8);
        v16h a;
#pragma unroll
        for (int i = 0; i < 8; ++i) { a[i] = a0[i]; a[8 + i] = a1[i]; }
        const int kb = k + (hi ? 16 : 0);
#pragma unroll
        for (int n = 0; n < 4; ++n) {
            const int col = n * 16 + l15;
            v16h b  = *(const v16h*)(WsT + (size_t)col * Kdim + kb);
            accS[n] = __builtin_amdgcn_wmma_f32_16x16x32_f16(false, a, false, b,
                                                             (short)0, accS[n], false, false);
            v16h b2 = *(const v16h*)(WdT + (size_t)col * Kdim + kb);
            accD[n] = __builtin_amdgcn_wmma_f32_16x16x32_f16(false, a, false, b2,
                                                             (short)0, accD[n], false, false);
        }
    }
    // C layout: VGPR v -> M = v (lanes 0-15) / v+8 (lanes 16-31); N = lane&15
#pragma unroll
    for (int n = 0; n < 4; ++n) {
        const int col = n * 16 + l15;
        const float bsv = bs[col], bdv = bd[col];
#pragma unroll
        for (int v = 0; v < 8; ++v) {
            const int row = rowBase + v + (hi ? 8 : 0);
            if (row < Nrows) {
                fs[(size_t)row * DCH + col] = accS[n][v] + bsv;
                fd[(size_t)row * DCH + col] = accD[n][v] + bdv;
            }
        }
    }
}

// ---------------- edge pass A: logits + segment-max (ordered-uint atomicMax) ----------------
__global__ void edge_logits_kernel(const float* __restrict__ fs, const float* __restrict__ fd,
                                   const float* __restrict__ attn,
                                   const int* __restrict__ src, const int* __restrict__ dst,
                                   int E, int H, int fshift,
                                   float* __restrict__ logits, unsigned* __restrict__ mmax) {
    __shared__ float sattn[DCH];
    if (threadIdx.x < DCH) sattn[threadIdx.x] = attn[threadIdx.x];
    __syncthreads();
    const int e = blockIdx.x * blockDim.x + threadIdx.x;
    if (e >= E) return;
    const int sN = src[e], dN = dst[e];
    const float* ps = fs + (size_t)sN * DCH;
    const float* pd = fd + (size_t)dN * DCH;
    float acc[8];
#pragma unroll
    for (int h = 0; h < 8; ++h) acc[h] = 0.f;
#pragma unroll
    for (int j = 0; j < DCH; j += 4) {
        float4 a = *(const float4*)(ps + j);
        float4 b = *(const float4*)(pd + j);
        const int h = j >> fshift;          // F>=8, so 4 consecutive feats share a head
        acc[h] += lrelu(a.x + b.x) * sattn[j]
                + lrelu(a.y + b.y) * sattn[j + 1]
                + lrelu(a.z + b.z) * sattn[j + 2]
                + lrelu(a.w + b.w) * sattn[j + 3];
    }
    for (int h = 0; h < H; ++h) {
        logits[(size_t)e * H + h] = acc[h];
        atomicMax(&mmax[(size_t)dN * H + h], fmap(acc[h]));
    }
}

// ---------------- edge pass B: pe = exp(logit - m) in-place, segment-sum ----------------
__global__ void edge_expsum_kernel(float* __restrict__ logits, const unsigned* __restrict__ mmax,
                                   float* __restrict__ ssum, const int* __restrict__ dst,
                                   int E, int H) {
    const long long idx = (long long)blockIdx.x * blockDim.x + threadIdx.x;
    if (idx >= (long long)E * H) return;
    const int e = (int)(idx / H), h = (int)(idx - (long long)e * H);
    const int dN = dst[e];
    const float m  = funmap(mmax[(size_t)dN * H + h]);
    const float pe = __expf(logits[idx] - m);
    logits[idx] = pe;
    atomicAdd(&ssum[(size_t)dN * H + h], pe);
}

// ---------------- edge pass C: out[dst] += (pe/s) * fs[src], 64 threads/edge ----------------
__global__ void edge_scatter_kernel(const float* __restrict__ pe, const float* __restrict__ ssum,
                                    const float* __restrict__ fs,
                                    const int* __restrict__ src, const int* __restrict__ dst,
                                    int E, int H, int fshift, float* __restrict__ acc) {
    const long long t = (long long)blockIdx.x * blockDim.x + threadIdx.x;
    const int e = (int)(t >> 6), j = (int)(t & 63);
    if (e >= E) return;
    const int sN = src[e], dN = dst[e];
    const int h = j >> fshift;
    const float coef = pe[(size_t)e * H + h] / ssum[(size_t)dN * H + h];
    atomicAdd(&acc[(size_t)dN * DCH + j], coef * fs[(size_t)sN * DCH + j]);
}

// ---------------- bias (+optional ELU), in place ----------------
__global__ void bias_act_kernel(float* __restrict__ h, const float* __restrict__ bias,
                                int n, int do_elu) {
    const long long i = (long long)blockIdx.x * blockDim.x + threadIdx.x;
    if (i >= (long long)n * DCH) return;
    const int c = (int)(i & 63);
    float v = h[i] + bias[c];
    if (do_elu) v = elu(v);
    h[i] = v;
}

// ---------------- BN stats: per-channel sum & sumsq (atomic) ----------------
__global__ void bn_stats_kernel(const float* __restrict__ h, int n, float* __restrict__ sums) {
    const int c = threadIdx.x & 63;
    const int rows_per_blk = blockDim.x >> 6;
    int r = blockIdx.x * rows_per_blk + (threadIdx.x >> 6);
    const int rstride = gridDim.x * rows_per_blk;
    float s = 0.f, q = 0.f;
    for (; r < n; r += rstride) {
        const float v = h[(size_t)r * DCH + c];
        s += v; q += v * v;
    }
    atomicAdd(&sums[c], s);
    atomicAdd(&sums[DCH + c], q);
}

// ---------------- BN apply (+optional ELU), writes f32 and/or f16 ----------------
__global__ void bn_apply_kernel(const float* __restrict__ h, const float* __restrict__ sums,
                                const float* __restrict__ gamma, const float* __restrict__ beta,
                                int n, int do_elu,
                                float* __restrict__ out32, _Float16* __restrict__ out16) {
    const long long i = (long long)blockIdx.x * blockDim.x + threadIdx.x;
    if (i >= (long long)n * DCH) return;
    const int c = (int)(i & 63);
    const float inv_n = 1.0f / (float)n;
    const float mu  = sums[c] * inv_n;
    const float var = sums[DCH + c] * inv_n - mu * mu;
    float v = (h[i] - mu) * rsqrtf(var + 1e-5f) * gamma[c] + beta[c];
    if (do_elu) v = elu(v);
    if (out32) out32[i] = v;
    if (out16) out16[i] = (_Float16)v;
}

// ---------------- host ----------------
extern "C" void kernel_launch(void* const* d_in, const int* in_sizes, int n_in,
                              void* d_out, int out_size, void* d_ws, size_t ws_size,
                              hipStream_t stream) {
    const float* x     = (const float*)d_in[0];
    const float* Wsrc1 = (const float*)d_in[1];  const float* bsrc1 = (const float*)d_in[2];
    const float* Wdst1 = (const float*)d_in[3];  const float* bdst1 = (const float*)d_in[4];
    const float* attn1 = (const float*)d_in[5];  const float* bias1 = (const float*)d_in[6];
    const float* gam1  = (const float*)d_in[7];  const float* bet1  = (const float*)d_in[8];
    const float* Wsrc2 = (const float*)d_in[9];  const float* bsrc2 = (const float*)d_in[10];
    const float* Wdst2 = (const float*)d_in[11]; const float* bdst2 = (const float*)d_in[12];
    const float* attn2 = (const float*)d_in[13]; const float* bias2 = (const float*)d_in[14];
    const float* gam2  = (const float*)d_in[15]; const float* bet2  = (const float*)d_in[16];
    const int*   src   = (const int*)d_in[17];   const int*   dst   = (const int*)d_in[18];

    const int N = in_sizes[0] / 128;
    const int E = in_sizes[17];
    const int IN_DIM = 128;

    // ---- carve workspace (256B aligned) ----
    char*  w   = (char*)d_ws;
    size_t off = 0;
    auto carve = [&](size_t bytes) { void* p = w + off; off += (bytes + 255) & ~(size_t)255; return p; };
    _Float16* x_h   = (_Float16*)carve((size_t)N * IN_DIM * 2);
    _Float16* h_h   = (_Float16*)carve((size_t)N * DCH * 2);
    _Float16* WsT1  = (_Float16*)carve((size_t)IN_DIM * DCH * 2);
    _Float16* WdT1  = (_Float16*)carve((size_t)IN_DIM * DCH * 2);
    _Float16* WsT2  = (_Float16*)carve((size_t)DCH * DCH * 2);
    _Float16* WdT2  = (_Float16*)carve((size_t)DCH * DCH * 2);
    float*    fsb   = (float*)carve((size_t)N * DCH * 4);
    float*    fdb   = (float*)carve((size_t)N * DCH * 4);
    float*    logb  = (float*)carve((size_t)E * 8 * 4);     // logits / pe (max H=8)
    unsigned* mmax  = (unsigned*)carve((size_t)N * 8 * 4);
    float*    ssum  = (float*)carve((size_t)N * 8 * 4);
    float*    accb  = (float*)carve((size_t)N * DCH * 4);
    float*    bnsum = (float*)carve(2 * DCH * 4);
    (void)ws_size; (void)n_in; (void)out_size;

    const int tiles  = (N + 15) / 16;
    const int gblk   = (tiles + 7) / 8;                      // 8 waves/block
    const int nEl    = N * DCH;
    const int nElBlk = (nEl + 255) / 256;
    const long long scat = (long long)E * 64;
    const int scatBlk = (int)((scat + 255) / 256);

    // ---- prep: pack weights, convert x ----
    pack_wt_kernel<<<(IN_DIM * DCH + 255) / 256, 256, 0, stream>>>(Wsrc1, IN_DIM, WsT1);
    pack_wt_kernel<<<(IN_DIM * DCH + 255) / 256, 256, 0, stream>>>(Wdst1, IN_DIM, WdT1);
    pack_wt_kernel<<<(DCH * DCH + 255) / 256, 256, 0, stream>>>(Wsrc2, DCH, WsT2);
    pack_wt_kernel<<<(DCH * DCH + 255) / 256, 256, 0, stream>>>(Wdst2, DCH, WdT2);
    cvt16_kernel<<<(N * IN_DIM + 255) / 256, 256, 0, stream>>>(x, x_h, (size_t)N * IN_DIM);

    // ================= layer 1 (H=8, F=8) =================
    gemm_dual_wmma_kernel<<<gblk, 256, 0, stream>>>(x_h, IN_DIM, N, WsT1, WdT1, bsrc1, bdst1, fsb, fdb);
    hipMemsetAsync(mmax, 0, (size_t)N * 8 * 4, stream);
    hipMemsetAsync(ssum, 0, (size_t)N * 8 * 4, stream);
    hipMemsetAsync(accb, 0, (size_t)N * DCH * 4, stream);
    edge_logits_kernel<<<(E + 255) / 256, 256, 0, stream>>>(fsb, fdb, attn1, src, dst, E, 8, 3, logb, mmax);
    edge_expsum_kernel<<<(E * 8 + 255) / 256, 256, 0, stream>>>(logb, mmax, ssum, dst, E, 8);
    edge_scatter_kernel<<<scatBlk, 256, 0, stream>>>(logb, ssum, fsb, src, dst, E, 8, 3, accb);
    bias_act_kernel<<<nElBlk, 256, 0, stream>>>(accb, bias1, N, 1);          // + bias1, ELU
    hipMemsetAsync(bnsum, 0, 2 * DCH * 4, stream);
    bn_stats_kernel<<<256, 256, 0, stream>>>(accb, N, bnsum);
    bn_apply_kernel<<<nElBlk, 256, 0, stream>>>(accb, bnsum, gam1, bet1, N, 1, nullptr, h_h); // BN, ELU -> f16

    // ================= layer 2 (H=4, F=16) =================
    gemm_dual_wmma_kernel<<<gblk, 256, 0, stream>>>(h_h, DCH, N, WsT2, WdT2, bsrc2, bdst2, fsb, fdb);
    hipMemsetAsync(mmax, 0, (size_t)N * 4 * 4, stream);
    hipMemsetAsync(ssum, 0, (size_t)N * 4 * 4, stream);
    hipMemsetAsync(accb, 0, (size_t)N * DCH * 4, stream);
    edge_logits_kernel<<<(E + 255) / 256, 256, 0, stream>>>(fsb, fdb, attn2, src, dst, E, 4, 4, logb, mmax);
    edge_expsum_kernel<<<(E * 4 + 255) / 256, 256, 0, stream>>>(logb, mmax, ssum, dst, E, 4);
    edge_scatter_kernel<<<scatBlk, 256, 0, stream>>>(logb, ssum, fsb, src, dst, E, 4, 4, accb);
    bias_act_kernel<<<nElBlk, 256, 0, stream>>>(accb, bias2, N, 0);          // + bias2, no ELU
    hipMemsetAsync(bnsum, 0, 2 * DCH * 4, stream);
    bn_stats_kernel<<<256, 256, 0, stream>>>(accb, N, bnsum);
    bn_apply_kernel<<<nElBlk, 256, 0, stream>>>(accb, bnsum, gam2, bet2, N, 0, (float*)d_out, nullptr);
}
